// EncoderBlock_26980984553867
// MI455X (gfx1250) — matrix operand order repaired and checked by reading, not compiled, not run on hardware
//
#include <hip/hip_runtime.h>

typedef __bf16 bf16;
typedef __attribute__((ext_vector_type(16))) __bf16 v16bf_t;
typedef __attribute__((ext_vector_type(8)))  __bf16 v8bf_t;
typedef __attribute__((ext_vector_type(8)))  float  v8f_t;
typedef __attribute__((ext_vector_type(4)))  unsigned int v4u_t;
typedef __attribute__((ext_vector_type(8)))  int v8i_t;
typedef __attribute__((ext_vector_type(4)))  int v4i_t;

static constexpr int D_MODEL  = 1024;
static constexpr int HEADS    = 16;
static constexpr int HEAD_DIM = 64;
static constexpr int DFF      = 4096;
static constexpr int BATCH    = 8;
static constexpr int SEQ      = 1024;
static constexpr int BS       = BATCH * SEQ;

#if __has_builtin(__builtin_amdgcn_tensor_load_to_lds)
#define USE_TDM 1
#else
#define USE_TDM 0
#endif

__device__ __forceinline__ v16bf_t comb16(v8bf_t lo, v8bf_t hi) {
  return __builtin_shufflevector(lo, hi, 0,1,2,3,4,5,6,7,8,9,10,11,12,13,14,15);
}

__device__ __forceinline__ v8f_t wmma_bf16f32(v16bf_t a, v16bf_t b, v8f_t c) {
  // D = A(16x32 bf16) x B(32x16 bf16) + C(16x16 f32)
  return __builtin_amdgcn_wmma_f32_16x16x32_bf16(false, a, false, b, (short)0, c, false, false);
}

#if USE_TDM
// Issue a TDM 2-D tile load: bf16 tile [tile_d1 rows][tile_d0 elems] from a
// row-major tensor with row stride `stride0` (elements) into LDS at ldsOff.
// D# packing per CDNA5 ISA; groups 2/3 and the
// trailing descriptor words are zero since the tensor is 2-D.
__device__ __forceinline__ void tdm_load_tile_2d(unsigned int ldsOff,
                                                 const bf16* gptr,
                                                 unsigned int tensor_d0,
                                                 unsigned int tensor_d1,
                                                 unsigned int tile_d0,
                                                 unsigned int tile_d1,
                                                 unsigned long long stride0) {
  unsigned long long ga = (unsigned long long)(uintptr_t)gptr;
  v4u_t g0;
  g0[0] = 1u;                                   // count=1, user descriptor
  g0[1] = ldsOff;                               // lds_addr (bytes)
  g0[2] = (unsigned int)ga;                     // global_addr[31:0]
  g0[3] = (unsigned int)((ga >> 32) & 0x1FFFFFFu) | (2u << 30); // addr[56:32] | type=2
  v8i_t g1;
  g1[0] = (int)(1u << 16);                      // data_size=1 -> 2 bytes/elem
  g1[1] = (int)((tensor_d0 & 0xFFFFu) << 16);   // tensor_dim0[15:0]
  g1[2] = (int)(((tensor_d0 >> 16) & 0xFFFFu) | ((tensor_d1 & 0xFFFFu) << 16));
  g1[3] = (int)(((tensor_d1 >> 16) & 0xFFFFu) | ((tile_d0 & 0xFFFFu) << 16));
  g1[4] = (int)(tile_d1 & 0xFFFFu);             // tile_dim1 (tile_dim2 = 0)
  g1[5] = (int)(unsigned int)(stride0 & 0xFFFFFFFFull);         // stride0[31:0]
  g1[6] = (int)(unsigned int)((stride0 >> 32) & 0xFFFFull);     // stride0[47:32]
  g1[7] = 0;                                    // stride1 unused (2-D)
  v4i_t z4 = {0, 0, 0, 0};
  v8i_t z8 = {0, 0, 0, 0, 0, 0, 0, 0};
  __builtin_amdgcn_tensor_load_to_lds(g0, g1, z4, z4, z8, 0);
}
#endif

// ---------------------------------------------------------------------------
// Weight convert + transpose: f32 in[K][N] -> bf16 out[N][K]
// ---------------------------------------------------------------------------
__global__ __launch_bounds__(256) void cvt_transpose_kernel(const float* __restrict__ in,
                                                            bf16* __restrict__ out,
                                                            int K, int N) {
  size_t i = (size_t)blockIdx.x * 256 + threadIdx.x;
  if (i >= (size_t)K * N) return;
  int n = (int)(i / K);
  int k = (int)(i % K);
  out[i] = (bf16)in[(size_t)k * N + n];
}

// ---------------------------------------------------------------------------
// LayerNorm over D_MODEL, one 256-thread workgroup per row, bf16 output
// ---------------------------------------------------------------------------
__global__ __launch_bounds__(256) void layernorm_kernel(const float* __restrict__ x,
                                                        const float* __restrict__ g,
                                                        const float* __restrict__ be,
                                                        bf16* __restrict__ out) {
  int row = blockIdx.x;
  int tid = threadIdx.x;
  const float* xr = x + (size_t)row * D_MODEL;
  float4 v = ((const float4*)xr)[tid];
  float s  = v.x + v.y + v.z + v.w;
  float s2 = v.x*v.x + v.y*v.y + v.z*v.z + v.w*v.w;
  #pragma unroll
  for (int off = 16; off >= 1; off >>= 1) {
    s  += __shfl_xor(s,  off, 32);
    s2 += __shfl_xor(s2, off, 32);
  }
  __shared__ float ws1[8], ws2[8];
  int lane = tid & 31, wave = tid >> 5;
  if (lane == 0) { ws1[wave] = s; ws2[wave] = s2; }
  __syncthreads();
  float tot = 0.f, tot2 = 0.f;
  #pragma unroll
  for (int i = 0; i < 8; ++i) { tot += ws1[i]; tot2 += ws2[i]; }
  float mu   = tot * (1.0f / D_MODEL);
  float var  = tot2 * (1.0f / D_MODEL) - mu * mu;
  float rstd = rsqrtf(var + 1e-6f);
  float4 gg = ((const float4*)g)[tid];
  float4 bb = ((const float4*)be)[tid];
  bf16* o = out + (size_t)row * D_MODEL + tid * 4;
  o[0] = (bf16)((v.x - mu) * rstd * gg.x + bb.x);
  o[1] = (bf16)((v.y - mu) * rstd * gg.y + bb.y);
  o[2] = (bf16)((v.z - mu) * rstd * gg.z + bb.z);
  o[3] = (bf16)((v.w - mu) * rstd * gg.w + bb.w);
}

// ---------------------------------------------------------------------------
// bf16 GEMM: C[M,N](f32 acc) = A[M,K] @ Bt[N,K]^T, fused bias/resid/relu.
// 128x128 tile per 256-thread workgroup, 8 waves each compute 64x32.
// Tiles staged to LDS by the Tensor Data Mover, double-buffered: wave 0
// DMAs the A tile, wave 1 the B tile, one K-chunk ahead of compute.
// ---------------------------------------------------------------------------
__global__ __launch_bounds__(256) void gemm_bf16_kernel(const bf16* __restrict__ A,
                                                        const bf16* __restrict__ Bt,
                                                        const float* __restrict__ bias,
                                                        const float* __restrict__ resid,
                                                        float* __restrict__ outF,
                                                        bf16* __restrict__ outB,
                                                        int M, int N, int K, int relu) {
  constexpr int TM = 128, TN = 128, TK = 32;
  __shared__ __align__(16) bf16 aLds[2][TM * TK];
  __shared__ __align__(16) bf16 bLds[2][TN * TK];
  const int tid  = threadIdx.x;
  const int lane = tid & 31;
  const int wave = tid >> 5;
  const int half = lane >> 4;
  const int ln16 = lane & 15;
  const int m0 = blockIdx.y * TM;
  const int n0 = blockIdx.x * TN;
  const int wm = (wave >> 2) * 64;   // wave row offset in tile
  const int wn = (wave & 3) * 32;    // wave col offset in tile

  const v8f_t vz = {0.f,0.f,0.f,0.f,0.f,0.f,0.f,0.f};
  v8f_t acc[4][2];
  #pragma unroll
  for (int rt = 0; rt < 4; ++rt) { acc[rt][0] = vz; acc[rt][1] = vz; }

  const int kIters = K / TK;

#if USE_TDM
  const unsigned int aOff[2] = {(unsigned int)(uintptr_t)&aLds[0][0],
                                (unsigned int)(uintptr_t)&aLds[1][0]};
  const unsigned int bOff[2] = {(unsigned int)(uintptr_t)&bLds[0][0],
                                (unsigned int)(uintptr_t)&bLds[1][0]};
  // Prologue: prefetch K-chunk 0 into buffer 0.
  if (wave == 0)
    tdm_load_tile_2d(aOff[0], A + (size_t)m0 * K, (unsigned)K, (unsigned)M,
                     TK, TM, (unsigned long long)K);
  if (wave == 1)
    tdm_load_tile_2d(bOff[0], Bt + (size_t)n0 * K, (unsigned)K, (unsigned)N,
                     TK, TN, (unsigned long long)K);
#endif

  for (int kb = 0; kb < kIters; ++kb) {
    const int cur = kb & 1;
#if USE_TDM
    if (wave == 0) {
      if (kb + 1 < kIters) {
        tdm_load_tile_2d(aOff[cur ^ 1], A + (size_t)m0 * K + (kb + 1) * TK,
                         (unsigned)K, (unsigned)M, TK, TM, (unsigned long long)K);
        __builtin_amdgcn_s_wait_tensorcnt(1);   // current tile done, prefetch in flight
      } else {
        __builtin_amdgcn_s_wait_tensorcnt(0);   // last tile: drain
      }
    }
    if (wave == 1) {
      if (kb + 1 < kIters) {
        tdm_load_tile_2d(bOff[cur ^ 1], Bt + (size_t)n0 * K + (kb + 1) * TK,
                         (unsigned)K, (unsigned)N, TK, TN, (unsigned long long)K);
        __builtin_amdgcn_s_wait_tensorcnt(1);
      } else {
        __builtin_amdgcn_s_wait_tensorcnt(0);
      }
    }
    __syncthreads();               // current buffer ready for all waves
#else
    const int k0 = kb * TK;
    __syncthreads();
    #pragma unroll
    for (int i = 0; i < 2; ++i) {
      int seg = tid + i * 256;     // 512 segments of 8 bf16
      int row = seg >> 2;
      int sc  = seg & 3;
      *(v8bf_t*)&aLds[cur][row * TK + sc * 8] =
          *(const v8bf_t*)&A[(size_t)(m0 + row) * K + k0 + sc * 8];
      *(v8bf_t*)&bLds[cur][row * TK + sc * 8] =
          *(const v8bf_t*)&Bt[(size_t)(n0 + row) * K + k0 + sc * 8];
    }
    __syncthreads();
#endif

    v16bf_t bfr[2];
    #pragma unroll
    for (int ct = 0; ct < 2; ++ct) {
      const bf16* p = &bLds[cur][(wn + ct * 16 + ln16) * TK];
      bfr[ct] = comb16(*(const v8bf_t*)&p[half * 8],
                       *(const v8bf_t*)&p[16 + half * 8]);
    }
    #pragma unroll
    for (int rt = 0; rt < 4; ++rt) {
      const bf16* p = &aLds[cur][(wm + rt * 16 + ln16) * TK];
      v16bf_t af = comb16(*(const v8bf_t*)&p[half * 8],
                          *(const v8bf_t*)&p[16 + half * 8]);
      acc[rt][0] = wmma_bf16f32(af, bfr[0], acc[rt][0]);
      acc[rt][1] = wmma_bf16f32(af, bfr[1], acc[rt][1]);
    }
#if USE_TDM
    __syncthreads();               // all reads of `cur` done before TDM rewrites it
#endif
  }

  float bv[2] = {0.f, 0.f};
  if (bias) {
    bv[0] = bias[n0 + wn + ln16];
    bv[1] = bias[n0 + wn + 16 + ln16];
  }
  #pragma unroll
  for (int rt = 0; rt < 4; ++rt) {
    #pragma unroll
    for (int ct = 0; ct < 2; ++ct) {
      int n = n0 + wn + ct * 16 + ln16;
      #pragma unroll
      for (int r = 0; r < 8; ++r) {
        int m = m0 + wm + rt * 16 + half * 8 + r;   // C layout: row = r + half*8
        float vv = acc[rt][ct][r] + bv[ct];
        size_t idx = (size_t)m * N + n;
        if (resid) vv += resid[idx];
        if (relu)  vv = fmaxf(vv, 0.f);
        if (outF)  outF[idx] = vv;
        if (outB)  outB[idx] = (bf16)vv;
      }
    }
  }
}

// ---------------------------------------------------------------------------
// Flash attention: block = 4 waves share one (b,h); each wave owns 16 q rows,
// online softmax over 32-key blocks; scores & P@V via bf16 WMMA.
// ---------------------------------------------------------------------------
__global__ __launch_bounds__(128) void attention_kernel(const bf16* __restrict__ Q,
                                                        const bf16* __restrict__ Km,
                                                        const bf16* __restrict__ V,
                                                        const unsigned char* __restrict__ mask,
                                                        bf16* __restrict__ Out) {
  __shared__ __align__(16) bf16 vt[HEAD_DIM * 32];   // V^T block: [dim][key]
  __shared__ __align__(16) bf16 pl[4][16 * 32];      // per-wave P tile [row][key]
  const int tid  = threadIdx.x;
  const int lane = tid & 31;
  const int wave = tid >> 5;
  const int half = lane >> 4;
  const int ln16 = lane & 15;
  const int b = blockIdx.z;
  const int h = blockIdx.y;
  const int q0 = blockIdx.x * 64 + wave * 16;
  const size_t base = (size_t)b * SEQ * D_MODEL + (size_t)h * HEAD_DIM;
  const unsigned char* mrow = mask + (size_t)b * SEQ;

  // Q fragments for the two d-chunks (A-matrix 16x32 layout)
  v16bf_t qf[2];
  #pragma unroll
  for (int c = 0; c < 2; ++c) {
    const bf16* p = &Q[base + (size_t)(q0 + ln16) * D_MODEL + c * 32];
    qf[c] = comb16(*(const v8bf_t*)&p[half * 8], *(const v8bf_t*)&p[16 + half * 8]);
  }

  const v8f_t vz = {0.f,0.f,0.f,0.f,0.f,0.f,0.f,0.f};
  float mI[8], lI[8];
  v8f_t o[4];
  #pragma unroll
  for (int r = 0; r < 8; ++r) { mI[r] = -1e30f; lI[r] = 0.f; }
  #pragma unroll
  for (int nt = 0; nt < 4; ++nt) o[nt] = vz;

  for (int j0 = 0; j0 < SEQ; j0 += 32) {
    __syncthreads();
    // Cooperative staging of V^T (32 keys x 64 dims -> vt[dim][key])
    #pragma unroll
    for (int i = 0; i < 2; ++i) {
      int seg = tid + i * 128;         // 256 segments of 8 dims
      int key = seg >> 3;
      int ds  = seg & 7;
      v8bf_t vv = *(const v8bf_t*)&V[base + (size_t)(j0 + key) * D_MODEL + ds * 8];
      #pragma unroll
      for (int e = 0; e < 8; ++e) vt[(ds * 8 + e) * 32 + key] = vv[e];
    }
    __syncthreads();

    // Scores: two 16x16 tiles, each over two K=32 d-chunks
    v8f_t s0 = vz, s1 = vz;
    #pragma unroll
    for (int c = 0; c < 2; ++c) {
      const bf16* p0 = &Km[base + (size_t)(j0 + ln16) * D_MODEL + c * 32];
      v16bf_t kf0 = comb16(*(const v8bf_t*)&p0[half * 8], *(const v8bf_t*)&p0[16 + half * 8]);
      s0 = wmma_bf16f32(qf[c], kf0, s0);
      const bf16* p1 = &Km[base + (size_t)(j0 + 16 + ln16) * D_MODEL + c * 32];
      v16bf_t kf1 = comb16(*(const v8bf_t*)&p1[half * 8], *(const v8bf_t*)&p1[16 + half * 8]);
      s1 = wmma_bf16f32(qf[c], kf1, s1);
    }
    const bool mk0 = mrow[j0 + ln16] != 0;        // key depends only on lane
    const bool mk1 = mrow[j0 + 16 + ln16] != 0;

    #pragma unroll
    for (int r = 0; r < 8; ++r) {
      float a0 = mk0 ? -1e30f : s0[r] * 0.125f;   // 1/sqrt(64)
      float a1 = mk1 ? -1e30f : s1[r] * 0.125f;
      float mx = fmaxf(a0, a1);
      #pragma unroll
      for (int off = 8; off >= 1; off >>= 1) mx = fmaxf(mx, __shfl_xor(mx, off, 32));
      float mnew = fmaxf(mI[r], mx);
      float scl  = __expf(mI[r] - mnew);
      float p0 = __expf(a0 - mnew);
      float p1 = __expf(a1 - mnew);
      float ps = p0 + p1;
      #pragma unroll
      for (int off = 8; off >= 1; off >>= 1) ps += __shfl_xor(ps, off, 32);
      lI[r] = lI[r] * scl + ps;
      mI[r] = mnew;
      #pragma unroll
      for (int nt = 0; nt < 4; ++nt) o[nt][r] = o[nt][r] * scl;
      // Write P (C layout -> LDS row-major) for re-layout into A fragment
      pl[wave][(half * 8 + r) * 32 + ln16]      = (bf16)p0;
      pl[wave][(half * 8 + r) * 32 + 16 + ln16] = (bf16)p1;
    }
    asm volatile("s_wait_dscnt 0" ::: "memory");   // order P store -> frag load (same wave)

    const bf16* pp = &pl[wave][ln16 * 32];
    v16bf_t pf = comb16(*(const v8bf_t*)&pp[half * 8], *(const v8bf_t*)&pp[16 + half * 8]);
    #pragma unroll
    for (int nt = 0; nt < 4; ++nt) {
      const bf16* vp = &vt[(nt * 16 + ln16) * 32];
      v16bf_t vf = comb16(*(const v8bf_t*)&vp[half * 8], *(const v8bf_t*)&vp[16 + half * 8]);
      o[nt] = wmma_bf16f32(pf, vf, o[nt]);
    }
  }

  #pragma unroll
  for (int r = 0; r < 8; ++r) {
    float inv = lI[r] > 0.f ? 1.f / lI[r] : 0.f;
    int m = q0 + half * 8 + r;
    #pragma unroll
    for (int nt = 0; nt < 4; ++nt)
      Out[base + (size_t)m * D_MODEL + nt * 16 + ln16] = (bf16)(o[nt][r] * inv);
  }
}

// ---------------------------------------------------------------------------
// Host orchestration
// ---------------------------------------------------------------------------
extern "C" void kernel_launch(void* const* d_in, const int* in_sizes, int n_in,
                              void* d_out, int out_size, void* d_ws, size_t ws_size,
                              hipStream_t stream) {
  (void)in_sizes; (void)n_in; (void)out_size; (void)ws_size;
  const float* x    = (const float*)d_in[0];
  const unsigned char* mask = (const unsigned char*)d_in[1];
  const float* Wq  = (const float*)d_in[2];
  const float* Wk  = (const float*)d_in[3];
  const float* Wv  = (const float*)d_in[4];
  const float* Wo  = (const float*)d_in[5];
  const float* W1  = (const float*)d_in[6];
  const float* b1  = (const float*)d_in[7];
  const float* W2  = (const float*)d_in[8];
  const float* b2  = (const float*)d_in[9];
  const float* g1  = (const float*)d_in[10];
  const float* be1 = (const float*)d_in[11];
  const float* g2  = (const float*)d_in[12];
  const float* be2 = (const float*)d_in[13];
  float* out = (float*)d_out;

  char* ws = (char*)d_ws;
  const size_t MB = 1ull << 20;
  bf16*  wqT = (bf16*)(ws + 0 * MB);     // 2 MB
  bf16*  wkT = (bf16*)(ws + 2 * MB);     // 2 MB
  bf16*  wvT = (bf16*)(ws + 4 * MB);     // 2 MB
  bf16*  woT = (bf16*)(ws + 6 * MB);     // 2 MB
  bf16*  w1T = (bf16*)(ws + 8 * MB);     // 8 MB  [DFF][D]
  bf16*  w2T = (bf16*)(ws + 16 * MB);    // 8 MB  [D][DFF]
  bf16*  h1  = (bf16*)(ws + 24 * MB);    // 16 MB LN1(x)
  bf16*  qb  = (bf16*)(ws + 40 * MB);    // 16 MB
  bf16*  kbf = (bf16*)(ws + 56 * MB);    // 16 MB
  bf16*  vbf = (bf16*)(ws + 72 * MB);    // 16 MB
  bf16*  ao  = (bf16*)(ws + 88 * MB);    // 16 MB attention out
  float* x1  = (float*)(ws + 104 * MB);  // 32 MB residual 1
  bf16*  h2  = (bf16*)(ws + 136 * MB);   // 16 MB LN2(x1)
  bf16*  f1  = (bf16*)(ws + 152 * MB);   // 64 MB relu(h2@W1+b1)

  // Weight bf16 transposes
  cvt_transpose_kernel<<<(D_MODEL * D_MODEL) / 256, 256, 0, stream>>>(Wq, wqT, D_MODEL, D_MODEL);
  cvt_transpose_kernel<<<(D_MODEL * D_MODEL) / 256, 256, 0, stream>>>(Wk, wkT, D_MODEL, D_MODEL);
  cvt_transpose_kernel<<<(D_MODEL * D_MODEL) / 256, 256, 0, stream>>>(Wv, wvT, D_MODEL, D_MODEL);
  cvt_transpose_kernel<<<(D_MODEL * D_MODEL) / 256, 256, 0, stream>>>(Wo, woT, D_MODEL, D_MODEL);
  cvt_transpose_kernel<<<(D_MODEL * DFF) / 256, 256, 0, stream>>>(W1, w1T, D_MODEL, DFF);
  cvt_transpose_kernel<<<(D_MODEL * DFF) / 256, 256, 0, stream>>>(W2, w2T, DFF, D_MODEL);

  // Sublayer 1: pre-norm MHA
  layernorm_kernel<<<BS, 256, 0, stream>>>(x, g1, be1, h1);

  dim3 gproj(D_MODEL / 128, BS / 128);
  gemm_bf16_kernel<<<gproj, 256, 0, stream>>>(h1, wqT, nullptr, nullptr, nullptr, qb,
                                              BS, D_MODEL, D_MODEL, 0);
  gemm_bf16_kernel<<<gproj, 256, 0, stream>>>(h1, wkT, nullptr, nullptr, nullptr, kbf,
                                              BS, D_MODEL, D_MODEL, 0);
  gemm_bf16_kernel<<<gproj, 256, 0, stream>>>(h1, wvT, nullptr, nullptr, nullptr, vbf,
                                              BS, D_MODEL, D_MODEL, 0);

  attention_kernel<<<dim3(SEQ / 64, HEADS, BATCH), 128, 0, stream>>>(qb, kbf, vbf, mask, ao);

  // x1 = x + attn @ Wo
  gemm_bf16_kernel<<<gproj, 256, 0, stream>>>(ao, woT, nullptr, x, x1, nullptr,
                                              BS, D_MODEL, D_MODEL, 0);

  // Sublayer 2: pre-norm FFN
  layernorm_kernel<<<BS, 256, 0, stream>>>(x1, g2, be2, h2);

  dim3 gff1(DFF / 128, BS / 128);
  gemm_bf16_kernel<<<gff1, 256, 0, stream>>>(h2, w1T, b1, nullptr, nullptr, f1,
                                             BS, DFF, D_MODEL, 1);

  // out = x1 + relu(...) @ W2 + b2
  gemm_bf16_kernel<<<gproj, 256, 0, stream>>>(f1, w2T, b2, x1, out, nullptr,
                                              BS, D_MODEL, DFF, 0);
}
